// GNNEncoder_65000035058026
// MI455X (gfx1250) — matrix operand order repaired and checked by reading, not compiled, not run on hardware
//
#include <hip/hip_runtime.h>
#include <hip/hip_bf16.h>

#define N_NODES 50000
#define HEADS   4
#define HID     64
#define NEG_SLOPE 0.2f
#define LN_EPS  1e-5f

typedef __attribute__((ext_vector_type(16))) __bf16 v16bf;
typedef __attribute__((ext_vector_type(8)))  float  v8f;

// ---------------- utility kernels ----------------

__global__ void k_f32_to_bf16(const float* __restrict__ in, __bf16* __restrict__ out, int n) {
    int i = blockIdx.x * blockDim.x + threadIdx.x;
    if (i < n) out[i] = (__bf16)in[i];
}

// W is [K, C] row-major -> Wt [C, K] bf16 (so B-operand K pairs are contiguous per column)
__global__ void k_transpose_bf16(const float* __restrict__ W, __bf16* __restrict__ Wt,
                                 int K, int C) {
    int i = blockIdx.x * blockDim.x + threadIdx.x;
    if (i < K * C) {
        int r = i / C, c = i - r * C;
        Wt[c * K + r] = (__bf16)W[i];
    }
}

__global__ void k_fill_f32(float* __restrict__ p, float v, int n) {
    int i = blockIdx.x * blockDim.x + threadIdx.x;
    if (i < n) p[i] = v;
}

__global__ void k_fill_i32(int* __restrict__ p, int v, int n) {
    int i = blockIdx.x * blockDim.x + threadIdx.x;
    if (i < n) p[i] = v;
}

// ---------------- WMMA GEMM ----------------
// C[M,Ncols] = A[M,K](bf16) * Bt[Ncols,K](bf16) (+bias)
// One wave computes a 16x64 output strip: 4 adjacent 16x16 tiles sharing the A fragment.
// K is compile-time (multiple of 32) -> fully unrolled, back-to-back v_wmma; straight-line
// fragment loads so the compiler coalesces each fragment into 2x global_load_b128.

template <int K, bool BIAS>
__global__ __launch_bounds__(256)
void k_wmma_gemm(const __bf16* __restrict__ A, const __bf16* __restrict__ Bt,
                 float* __restrict__ C, const float* __restrict__ bias,
                 int M, int Ncols) {
    int wave = (blockIdx.x * blockDim.x + threadIdx.x) >> 5;
    int lane = threadIdx.x & 31;
    int groupsN = Ncols >> 6;                   // 4 tiles (64 cols) per wave
    int waves   = (M >> 4) * groupsN;
    if (wave >= waves) return;                  // wave-uniform: EXEC stays all-ones for WMMA
    int tm = wave / groupsN;
    int tg = wave - tm * groupsN;

    int l15 = lane & 15;
    int hi  = (lane >> 4) & 1;
    int m  = tm * 16 + l15;                     // A row for this lane
    int n0 = tg * 64 + l15;                     // first B column for this lane

    const unsigned int* A32 = (const unsigned int*)A;
    const unsigned int* B32 = (const unsigned int*)Bt;

    v8f acc0 = {}, acc1 = {}, acc2 = {}, acc3 = {};
#pragma unroll
    for (int k0 = 0; k0 < K; k0 += 32) {
        union { unsigned int u[8]; v16bf v; } a, b0, b1, b2, b3;
#pragma unroll
        for (int r = 0; r < 8; ++r) {
            // A 16x32 bf16 layout: VGPR r -> K = 16*(r>=4) + 2*(r&3) + 8*hi  (pair K,K+1)
            int kbA = ((r & 4) << 2) + ((r & 3) << 1) + (hi << 3);
            // B 32x16 bf16 layout: same K order as A: VGPR r -> K = 16*(r>=4) + 2*(r&3) + 8*hi
            int kbB = ((r & 4) << 2) + ((r & 3) << 1) + (hi << 3);
            a.u[r]  = A32[(m * K + k0 + kbA) >> 1];
            b0.u[r] = B32[((n0 +  0) * K + k0 + kbB) >> 1];
            b1.u[r] = B32[((n0 + 16) * K + k0 + kbB) >> 1];
            b2.u[r] = B32[((n0 + 32) * K + k0 + kbB) >> 1];
            b3.u[r] = B32[((n0 + 48) * K + k0 + kbB) >> 1];
        }
        acc0 = __builtin_amdgcn_wmma_f32_16x16x32_bf16(false, a.v, false, b0.v, (short)0, acc0, false, false);
        acc1 = __builtin_amdgcn_wmma_f32_16x16x32_bf16(false, a.v, false, b1.v, (short)0, acc1, false, false);
        acc2 = __builtin_amdgcn_wmma_f32_16x16x32_bf16(false, a.v, false, b2.v, (short)0, acc2, false, false);
        acc3 = __builtin_amdgcn_wmma_f32_16x16x32_bf16(false, a.v, false, b3.v, (short)0, acc3, false, false);
    }

    int rowBase = tm * 16 + (hi << 3);
#pragma unroll
    for (int j = 0; j < 4; ++j) {
        const v8f& acc = (j == 0) ? acc0 : (j == 1) ? acc1 : (j == 2) ? acc2 : acc3;
        int nj = n0 + 16 * j;
        float bv = BIAS ? bias[nj] : 0.f;
#pragma unroll
        for (int i = 0; i < 8; ++i) {
            // C VGPR i: M=i (lanes 0-15) / M=8+i (lanes 16-31), N = lane&15 within tile
            C[(rowBase + i) * Ncols + nj] = acc[i] + bv;
        }
    }
}

// ---------------- attention logits: es/ed[N,H] = sum_c H[n,h,c] * a{src,dst}[h,c] ----------------

__global__ void k_alphas(const float* __restrict__ H, const float* __restrict__ asrc,
                         const float* __restrict__ adst, float* __restrict__ es,
                         float* __restrict__ ed, int Nn) {
    int i = blockIdx.x * blockDim.x + threadIdx.x;
    if (i >= Nn * HEADS) return;
    int nd = i >> 2, hh = i & 3;
    const float* hp = H + nd * (HEADS * HID) + hh * HID;
    const float* as = asrc + hh * HID;
    const float* ad = adst + hh * HID;
    float s = 0.f, d = 0.f;
#pragma unroll 8
    for (int c = 0; c < HID; ++c) { s += hp[c] * as[c]; d += hp[c] * ad[c]; }
    es[i] = s; ed[i] = d;
}

// ---------------- edge passes ----------------

__device__ __forceinline__ int f2ord(float f) {
    int i = __float_as_int(f);
    return i >= 0 ? i : (i ^ 0x7fffffff);
}
__device__ __forceinline__ float ord2f(int o) {
    return __int_as_float(o >= 0 ? o : (o ^ 0x7fffffff));
}

__device__ __forceinline__ void edge_sd(const int* sI, const int* dI, int Eorig, int e,
                                        int& s, int& d) {
    if (e < Eorig) { s = sI[e]; d = dI[e]; }
    else           { s = e - Eorig; d = s; }  // self loops appended
}

__global__ void k_edge_max(const int* __restrict__ sI, const int* __restrict__ dI,
                           int Eorig, int Etot, const float* __restrict__ es,
                           const float* __restrict__ ed, int* __restrict__ mx) {
    int i = blockIdx.x * blockDim.x + threadIdx.x;
    if (i >= Etot * HEADS) return;
    int e = i >> 2, hh = i & 3, s, d;
    edge_sd(sI, dI, Eorig, e, s, d);
    float v = es[s * HEADS + hh] + ed[d * HEADS + hh];
    v = v > 0.f ? v : v * NEG_SLOPE;
    atomicMax(&mx[d * HEADS + hh], f2ord(v));
}

__global__ void k_edge_exp(const int* __restrict__ sI, const int* __restrict__ dI,
                           int Eorig, int Etot, const float* __restrict__ es,
                           const float* __restrict__ ed, const int* __restrict__ mx,
                           float* __restrict__ ex, float* __restrict__ den) {
    int i = blockIdx.x * blockDim.x + threadIdx.x;
    if (i >= Etot * HEADS) return;
    int e = i >> 2, hh = i & 3, s, d;
    edge_sd(sI, dI, Eorig, e, s, d);
    float v = es[s * HEADS + hh] + ed[d * HEADS + hh];
    v = v > 0.f ? v : v * NEG_SLOPE;
    float xv = __expf(v - ord2f(mx[d * HEADS + hh]));
    ex[i] = xv;
    atomicAdd(&den[d * HEADS + hh], xv);
}

// one block per edge, 256 threads = (head, channel); acc[d, h*64+c] += H[s, h*64+c] * alpha
__global__ __launch_bounds__(256)
void k_edge_msg(const int* __restrict__ sI, const int* __restrict__ dI, int Eorig,
                const float* __restrict__ H, const float* __restrict__ ex,
                const float* __restrict__ den, float* __restrict__ acc) {
    int e = blockIdx.x;
    int t = threadIdx.x;          // h*HID + c
    int hh = t >> 6;
    int s, d;
    edge_sd(sI, dI, Eorig, e, s, d);
    float alpha = ex[e * HEADS + hh] / den[d * HEADS + hh];
    atomicAdd(&acc[d * (HEADS * HID) + t], H[s * (HEADS * HID) + t] * alpha);
}

// ---------------- head-mean + bias + LayerNorm + ReLU ----------------
// one block of 64 threads per node

__global__ __launch_bounds__(64)
void k_node_finish(const float* __restrict__ acc, const float* __restrict__ b,
                   const float* __restrict__ g, const float* __restrict__ be,
                   float* __restrict__ out) {
    int nd = blockIdx.x;
    int c  = threadIdx.x;
    const float* ap = acc + nd * (HEADS * HID);
    float v = 0.25f * (ap[c] + ap[HID + c] + ap[2 * HID + c] + ap[3 * HID + c]) + b[c];
    __shared__ float sv[HID];
    __shared__ float mean_s, rstd_s;
    sv[c] = v;
    __syncthreads();
    if (c == 0) {
        float s = 0.f;
        for (int i = 0; i < HID; ++i) s += sv[i];
        float m = s * (1.f / HID);
        float vv = 0.f;
        for (int i = 0; i < HID; ++i) { float df = sv[i] - m; vv += df * df; }
        mean_s = m;
        rstd_s = rsqrtf(vv * (1.f / HID) + LN_EPS);
    }
    __syncthreads();
    float o = (v - mean_s) * rstd_s * g[c] + be[c];
    out[nd * HID + c] = fmaxf(o, 0.f);
}

// ---------------- host side ----------------

extern "C" void kernel_launch(void* const* d_in, const int* in_sizes, int n_in,
                              void* d_out, int out_size, void* d_ws, size_t ws_size,
                              hipStream_t stream) {
    const float* x     = (const float*)d_in[0];
    const int*   eidx  = (const int*)  d_in[1];
    const float* W1    = (const float*)d_in[2];
    const float* asrc1 = (const float*)d_in[3];
    const float* adst1 = (const float*)d_in[4];
    const float* b1    = (const float*)d_in[5];
    const float* g1    = (const float*)d_in[6];
    const float* be1   = (const float*)d_in[7];
    const float* W2    = (const float*)d_in[8];
    const float* asrc2 = (const float*)d_in[9];
    const float* adst2 = (const float*)d_in[10];
    const float* b2    = (const float*)d_in[11];
    const float* g2    = (const float*)d_in[12];
    const float* be2   = (const float*)d_in[13];
    const float* Wo    = (const float*)d_in[14];
    const float* bo    = (const float*)d_in[15];

    const int Nn    = N_NODES;
    const int IN_C  = 128;
    const int OUT_C = 128;
    const int E     = in_sizes[1] / 2;
    const int Etot  = E + Nn;
    const int* srcI = eidx;
    const int* dstI = eidx + E;

    // carve workspace
    char* ws = (char*)d_ws;
    size_t off = 0;
    auto take = [&](size_t bytes) -> char* {
        char* p = ws + off;
        off = (off + bytes + 255) & ~(size_t)255;
        return p;
    };
    float*  H   = (float*) take((size_t)Nn * HEADS * HID * 4);   // GEMM result [N,256]
    float*  ACC = (float*) take((size_t)Nn * HEADS * HID * 4);   // message accumulation
    __bf16* XB  = (__bf16*)take((size_t)Nn * IN_C * 2);          // bf16 layer input
    __bf16* WT  = (__bf16*)take((size_t)256 * 128 * 2);          // bf16 transposed weights
    float*  ES  = (float*) take((size_t)Nn * HEADS * 4);
    float*  ED  = (float*) take((size_t)Nn * HEADS * 4);
    int*    MX  = (int*)   take((size_t)Nn * HEADS * 4);
    float*  DEN = (float*) take((size_t)Nn * HEADS * 4);
    float*  EX  = (float*) take((size_t)Etot * HEADS * 4);
    float*  HN  = (float*) take((size_t)Nn * HID * 4);           // per-layer node output
    (void)ws_size; (void)n_in; (void)out_size;

    const int TB = 256;
    auto blocks = [](int n, int tb) { return (n + tb - 1) / tb; };

    auto edge_softmax_scatter = [&](const float* asrc, const float* adst,
                                    const float* bb, const float* gg, const float* bbe) {
        k_alphas<<<blocks(Nn * HEADS, TB), TB, 0, stream>>>(H, asrc, adst, ES, ED, Nn);
        k_fill_i32<<<blocks(Nn * HEADS, TB), TB, 0, stream>>>(MX, (int)0x80000000, Nn * HEADS);
        k_fill_f32<<<blocks(Nn * HEADS, TB), TB, 0, stream>>>(DEN, 0.f, Nn * HEADS);
        k_fill_f32<<<blocks(Nn * HEADS * HID, TB), TB, 0, stream>>>(ACC, 0.f, Nn * HEADS * HID);
        k_edge_max<<<blocks(Etot * HEADS, TB), TB, 0, stream>>>(srcI, dstI, E, Etot, ES, ED, MX);
        k_edge_exp<<<blocks(Etot * HEADS, TB), TB, 0, stream>>>(srcI, dstI, E, Etot, ES, ED,
                                                                MX, EX, DEN);
        k_edge_msg<<<Etot, 256, 0, stream>>>(srcI, dstI, E, H, EX, DEN, ACC);
        k_node_finish<<<Nn, 64, 0, stream>>>(ACC, bb, gg, bbe, HN);
    };

    // ---- layer 1: H = bf16(x) @ W1 ----
    k_f32_to_bf16<<<blocks(Nn * IN_C, TB), TB, 0, stream>>>(x, XB, Nn * IN_C);
    k_transpose_bf16<<<blocks(128 * 256, TB), TB, 0, stream>>>(W1, WT, 128, 256);
    {
        int waves = (Nn >> 4) * (256 >> 6);
        k_wmma_gemm<128, false><<<blocks(waves * 32, TB), TB, 0, stream>>>(XB, WT, H, nullptr,
                                                                           Nn, 256);
    }
    edge_softmax_scatter(asrc1, adst1, b1, g1, be1);

    // ---- layer 2: H = bf16(HN) @ W2 ----
    k_f32_to_bf16<<<blocks(Nn * HID, TB), TB, 0, stream>>>(HN, XB, Nn * HID);
    k_transpose_bf16<<<blocks(HID * 256, TB), TB, 0, stream>>>(W2, WT, HID, 256);
    {
        int waves = (Nn >> 4) * (256 >> 6);
        k_wmma_gemm<HID, false><<<blocks(waves * 32, TB), TB, 0, stream>>>(XB, WT, H, nullptr,
                                                                           Nn, 256);
    }
    edge_softmax_scatter(asrc2, adst2, b2, g2, be2);

    // ---- output projection: d_out = bf16(HN) @ Wo + bo ----
    k_f32_to_bf16<<<blocks(Nn * HID, TB), TB, 0, stream>>>(HN, XB, Nn * HID);
    k_transpose_bf16<<<blocks(HID * OUT_C, TB), TB, 0, stream>>>(Wo, WT, HID, OUT_C);
    {
        int waves = (Nn >> 4) * (OUT_C >> 6);
        k_wmma_gemm<HID, true><<<blocks(waves * 32, TB), TB, 0, stream>>>(XB, WT, (float*)d_out,
                                                                          bo, Nn, OUT_C);
    }
}
